// SoftAttention_51668456571095
// MI455X (gfx1250) — matrix operand order repaired and checked by reading, not compiled, not run on hardware
//
#include <hip/hip_runtime.h>
#include <hip/hip_bf16.h>

// ---------------------------------------------------------------------------
// Problem constants (B=256, T=128, H=1024, E=1024)
// ---------------------------------------------------------------------------
#define BB   256
#define TT   128
#define HH   1024
#define EE   1024
#define BT   (BB * TT)          // 32768 rows for the big GEMMs

// WMMA tiling: 128x128 C tile per block, K staged 32 at a time.
// 8 waves (256 threads): wave grid 4 (M) x 2 (N); each wave owns 32x64,
// i.e. 2x4 = 8 accumulator tiles of 16x16.
#define LDSTRIDE 56             // bf16 elems per LDS row (112B: 16B-aligned, bank-clean)

typedef __attribute__((ext_vector_type(8)))  __bf16 v8bf;
typedef __attribute__((ext_vector_type(16))) __bf16 v16bf;
typedef __attribute__((ext_vector_type(8)))  float  v8f;

union Frag { v16bf v; v8bf h[2]; };

// ---------------------------------------------------------------------------
// Fragment loads from LDS, per CDNA5 ISA VGPR layouts (CDNA5 ISA §7.12.2):
//   A 16x32 bf16: lane m<16 holds K {0..7,16..23}; lane m+16 holds K {8..15,24..31}
//   B 32x16 bf16: lane n<16 holds K {0..7,16..23}; lane n+16 holds K {8..15,24..31}
// ---------------------------------------------------------------------------
__device__ __forceinline__ v16bf load_a_frag(const __bf16* tile, int lane) {
  const int m  = lane & 15;
  const int kh = (lane >> 4) & 1;
  const __bf16* p = tile + m * LDSTRIDE + kh * 8;
  Frag f;
  f.h[0] = *(const v8bf*)(p);
  f.h[1] = *(const v8bf*)(p + 16);
  return f.v;
}

__device__ __forceinline__ v16bf load_b_frag(const __bf16* tile, int lane) {
  const int n  = lane & 15;
  const int kh = (lane >> 4) & 1;
  const __bf16* p = tile + n * LDSTRIDE + kh * 8;
  Frag f;
  f.h[0] = *(const v8bf*)(p);
  f.h[1] = *(const v8bf*)(p + 16);
  return f.v;
}

// Convert 16 f32 (held in 4 float4 regs) into bf16 hi/lo pairs, store to LDS.
__device__ __forceinline__ void cvt_store(const float4* q, __bf16* dh, __bf16* dl) {
  float f[16];
  *(float4*)(f + 0)  = q[0]; *(float4*)(f + 4)  = q[1];
  *(float4*)(f + 8)  = q[2]; *(float4*)(f + 12) = q[3];
  v8bf h0, l0, h1, l1;
#pragma unroll
  for (int i = 0; i < 8; ++i) {
    __bf16 hi = (__bf16)f[i];
    h0[i] = hi; l0[i] = (__bf16)(f[i] - (float)hi);
  }
#pragma unroll
  for (int i = 0; i < 8; ++i) {
    __bf16 hi = (__bf16)f[8 + i];
    h1[i] = hi; l1[i] = (__bf16)(f[8 + i] - (float)hi);
  }
  *(v8bf*)(dh)     = h0; *(v8bf*)(dh + 8) = h1;
  *(v8bf*)(dl)     = l0; *(v8bf*)(dl + 8) = l1;
}

__device__ __forceinline__ void gload16(const float* __restrict__ g, float4* q) {
  const float4* g4 = (const float4*)g;
  q[0] = g4[0]; q[1] = g4[1]; q[2] = g4[2]; q[3] = g4[3];
}

__device__ __forceinline__ float tanh_fast(float x) {
  float xc = fminf(fmaxf(x, -15.f), 15.f);
  float e  = __expf(2.f * xc);          // v_exp_f32 path
  return (e - 1.f) / (e + 1.f);
}

#define WMMA_BF16(A, B, C) \
  __builtin_amdgcn_wmma_f32_16x16x32_bf16(false, (A), false, (B), (short)0, (C), false, false)

// ---------------------------------------------------------------------------
// Kernel 1: partial attention scores.
//   grid = (256 M-tiles, 16 sections); section = {mat in 0..1} x {ntile in 0..7}
//   For its 128x128 tile of hx (mat=0) or vx (mat=1), computes
//   sum_n tanh(tile[m,n]) * weight[off + n]  and stores per-(section,wn) row
//   partials to ws (deterministic: no atomics).
//   Register-prefetch pipelining: next K-step's global tile is loaded into
//   VGPRs while the current step's 24 WMMAs run from LDS.
// ---------------------------------------------------------------------------
__global__ __launch_bounds__(256)
void k_scores(const float* __restrict__ h, const float* __restrict__ aspect,
              const float* __restrict__ Wh, const float* __restrict__ Wv,
              const float* __restrict__ weight, float* __restrict__ part) {
  const int mtile = blockIdx.x;           // 0..255
  const int sec   = blockIdx.y;           // 0..15
  const int mat   = sec >> 3;             // 0: h*Wh, 1: aspect*Wv
  const int nt    = sec & 7;              // N tile within the 1024 outputs

  const float* Aptr = mat ? aspect : h;
  const float* Bptr = mat ? Wv : Wh;
  const int    woff = mat ? HH : 0;
  const int    Arow0 = mtile * 128;
  const int    Ncol0 = nt * 128;

  __shared__ __attribute__((aligned(16))) __bf16 Ah[128 * LDSTRIDE];
  __shared__ __attribute__((aligned(16))) __bf16 Al[128 * LDSTRIDE];
  __shared__ __attribute__((aligned(16))) __bf16 Bh[128 * LDSTRIDE];
  __shared__ __attribute__((aligned(16))) __bf16 Bl[128 * LDSTRIDE];

  const int tid  = threadIdx.x;
  const int lane = tid & 31;
  const int wave = tid >> 5;
  const int wm   = wave & 3;              // 4 M groups of 32 rows
  const int wn   = wave >> 2;             // 2 N groups of 64 cols
  const int srow = tid >> 1;              // staging row 0..127
  const int scol = (tid & 1) << 4;        // staging col 0 or 16

  const float* gA = Aptr + (size_t)(Arow0 + srow) * HH + scol;
  const float* gB = Bptr + (size_t)(Ncol0 + srow) * HH + scol;

  v8f acc[8];
  v8f zero = {0.f, 0.f, 0.f, 0.f, 0.f, 0.f, 0.f, 0.f};
#pragma unroll
  for (int i = 0; i < 8; ++i) acc[i] = zero;

  float4 pa[4], pb[4];
  gload16(gA, pa);                        // prefetch K-step 0
  gload16(gB, pb);

#pragma unroll 1
  for (int kk = 0; kk < HH / 32; ++kk) {
    __syncthreads();                      // previous compute done reading LDS
    cvt_store(pa, &Ah[srow * LDSTRIDE + scol], &Al[srow * LDSTRIDE + scol]);
    cvt_store(pb, &Bh[srow * LDSTRIDE + scol], &Bl[srow * LDSTRIDE + scol]);
    __syncthreads();

    if (kk + 1 < HH / 32) {               // prefetch next K-step (overlaps WMMAs)
      gload16(gA + (kk + 1) * 32, pa);
      gload16(gB + (kk + 1) * 32, pb);
    }

    v16bf ah[2], al[2], bh[4], bl[4];
#pragma unroll
    for (int ms = 0; ms < 2; ++ms) {
      ah[ms] = load_a_frag(&Ah[(wm * 32 + ms * 16) * LDSTRIDE], lane);
      al[ms] = load_a_frag(&Al[(wm * 32 + ms * 16) * LDSTRIDE], lane);
    }
#pragma unroll
    for (int ns = 0; ns < 4; ++ns) {
      bh[ns] = load_b_frag(&Bh[(wn * 64 + ns * 16) * LDSTRIDE], lane);
      bl[ns] = load_b_frag(&Bl[(wn * 64 + ns * 16) * LDSTRIDE], lane);
    }
#pragma unroll
    for (int ms = 0; ms < 2; ++ms)
#pragma unroll
      for (int ns = 0; ns < 4; ++ns) {
        const int t = ms * 4 + ns;
        acc[t] = WMMA_BF16(ah[ms], bh[ns], acc[t]);   // hi*hi
        acc[t] = WMMA_BF16(ah[ms], bl[ns], acc[t]);   // hi*lo
        acc[t] = WMMA_BF16(al[ms], bh[ns], acc[t]);   // lo*hi
      }
  }

  // Epilogue: score partial = sum_n tanh(v) * weight[woff + ncol]
  float wv[4];
#pragma unroll
  for (int ns = 0; ns < 4; ++ns)
    wv[ns] = weight[woff + Ncol0 + wn * 64 + ns * 16 + (lane & 15)];

  const int slot = sec * 2 + wn;          // 0..31 deterministic slices
#pragma unroll
  for (int ms = 0; ms < 2; ++ms) {
#pragma unroll
    for (int i = 0; i < 8; ++i) {
      float s = 0.f;
#pragma unroll
      for (int ns = 0; ns < 4; ++ns)
        s += tanh_fast(acc[ms * 4 + ns][i]) * wv[ns];
      // reduce the 16 N-columns held across each 16-lane half
      s += __shfl_xor(s, 1, 32);
      s += __shfl_xor(s, 2, 32);
      s += __shfl_xor(s, 4, 32);
      s += __shfl_xor(s, 8, 32);
      if ((lane & 15) == 0) {
        const int row = Arow0 + wm * 32 + ms * 16 + i + ((lane >> 4) << 3);
        part[(size_t)slot * BT + row] = s;
      }
    }
  }
}

// ---------------------------------------------------------------------------
// Kernel 2: reduce partials -> softmax over T -> rx = ax^T h ; pack [rx || hn]
// ---------------------------------------------------------------------------
__global__ __launch_bounds__(256)
void k_softmax_rx(const float* __restrict__ part, const float* __restrict__ h,
                  float* __restrict__ Araw) {
  const int b = blockIdx.x;
  const int t = threadIdx.x;
  __shared__ float sc[TT];
  __shared__ float ax[TT];

  if (t < TT) {
    float s = 0.f;
#pragma unroll
    for (int j = 0; j < 32; ++j) s += part[(size_t)j * BT + b * TT + t];
    sc[t] = s;
  }
  __syncthreads();

  if (t < 32) {
    float v0 = sc[t * 4 + 0], v1 = sc[t * 4 + 1];
    float v2 = sc[t * 4 + 2], v3 = sc[t * 4 + 3];
    float m = fmaxf(fmaxf(v0, v1), fmaxf(v2, v3));
    for (int off = 1; off < 32; off <<= 1) m = fmaxf(m, __shfl_xor(m, off, 32));
    float e0 = __expf(v0 - m), e1 = __expf(v1 - m);
    float e2 = __expf(v2 - m), e3 = __expf(v3 - m);
    float sum = e0 + e1 + e2 + e3;
    for (int off = 1; off < 32; off <<= 1) sum += __shfl_xor(sum, off, 32);
    float inv = 1.f / sum;
    ax[t * 4 + 0] = e0 * inv; ax[t * 4 + 1] = e1 * inv;
    ax[t * 4 + 2] = e2 * inv; ax[t * 4 + 3] = e3 * inv;
  }
  __syncthreads();

  const float* hb = h + (size_t)b * TT * HH;
  const int col = t * 4;
  float rx0 = 0.f, rx1 = 0.f, rx2 = 0.f, rx3 = 0.f;
  for (int tt = 0; tt < TT; ++tt) {
    const float a = ax[tt];
    float4 v = *(const float4*)(hb + (size_t)tt * HH + col);
    rx0 = fmaf(a, v.x, rx0); rx1 = fmaf(a, v.y, rx1);
    rx2 = fmaf(a, v.z, rx2); rx3 = fmaf(a, v.w, rx3);
  }
  float4 r = {rx0, rx1, rx2, rx3};
  *(float4*)(Araw + (size_t)b * 2048 + col) = r;
  float4 hn = *(const float4*)(hb + (size_t)(TT - 1) * HH + col);
  *(float4*)(Araw + (size_t)b * 2048 + HH + col) = hn;
}

// ---------------------------------------------------------------------------
// Kernel 3: hs = tanh([rx||hn] (256x2048) x [Wp;Wx]^T (2048x1024))
//   grid = (2 M-tiles, 8 N-tiles), same bf16x3 WMMA tile + register prefetch.
// ---------------------------------------------------------------------------
__global__ __launch_bounds__(256)
void k_final(const float* __restrict__ Araw, const float* __restrict__ Wp,
             const float* __restrict__ Wx, float* __restrict__ out) {
  const int Arow0 = blockIdx.x * 128;     // 0 or 128
  const int Ncol0 = blockIdx.y * 128;     // 0..896

  __shared__ __attribute__((aligned(16))) __bf16 Ah[128 * LDSTRIDE];
  __shared__ __attribute__((aligned(16))) __bf16 Al[128 * LDSTRIDE];
  __shared__ __attribute__((aligned(16))) __bf16 Bh[128 * LDSTRIDE];
  __shared__ __attribute__((aligned(16))) __bf16 Bl[128 * LDSTRIDE];

  const int tid  = threadIdx.x;
  const int lane = tid & 31;
  const int wave = tid >> 5;
  const int wm   = wave & 3;
  const int wn   = wave >> 2;
  const int srow = tid >> 1;
  const int scol = (tid & 1) << 4;

  v8f acc[8];
  v8f zero = {0.f, 0.f, 0.f, 0.f, 0.f, 0.f, 0.f, 0.f};
#pragma unroll
  for (int i = 0; i < 8; ++i) acc[i] = zero;

  const float* gA = Araw + (size_t)(Arow0 + srow) * 2048 + scol;
  // B source for a given k-chunk (16-wide chunk never straddles HH)
  auto bptr = [&](int kg) -> const float* {
    return (kg < HH) ? (Wp + (size_t)(Ncol0 + srow) * HH + kg)
                     : (Wx + (size_t)(Ncol0 + srow) * HH + (kg - HH));
  };

  float4 pa[4], pb[4];
  gload16(gA, pa);
  gload16(bptr(scol), pb);

#pragma unroll 1
  for (int kk = 0; kk < 2048 / 32; ++kk) {
    __syncthreads();
    cvt_store(pa, &Ah[srow * LDSTRIDE + scol], &Al[srow * LDSTRIDE + scol]);
    cvt_store(pb, &Bh[srow * LDSTRIDE + scol], &Bl[srow * LDSTRIDE + scol]);
    __syncthreads();

    if (kk + 1 < 2048 / 32) {
      gload16(gA + (kk + 1) * 32, pa);
      gload16(bptr((kk + 1) * 32 + scol), pb);
    }

    v16bf ah[2], al[2], bh[4], bl[4];
#pragma unroll
    for (int ms = 0; ms < 2; ++ms) {
      ah[ms] = load_a_frag(&Ah[(wm * 32 + ms * 16) * LDSTRIDE], lane);
      al[ms] = load_a_frag(&Al[(wm * 32 + ms * 16) * LDSTRIDE], lane);
    }
#pragma unroll
    for (int ns = 0; ns < 4; ++ns) {
      bh[ns] = load_b_frag(&Bh[(wn * 64 + ns * 16) * LDSTRIDE], lane);
      bl[ns] = load_b_frag(&Bl[(wn * 64 + ns * 16) * LDSTRIDE], lane);
    }
#pragma unroll
    for (int ms = 0; ms < 2; ++ms)
#pragma unroll
      for (int ns = 0; ns < 4; ++ns) {
        const int t = ms * 4 + ns;
        acc[t] = WMMA_BF16(ah[ms], bh[ns], acc[t]);
        acc[t] = WMMA_BF16(ah[ms], bl[ns], acc[t]);
        acc[t] = WMMA_BF16(al[ms], bh[ns], acc[t]);
      }
  }

  // Epilogue: hs = tanh(acc)
#pragma unroll
  for (int ms = 0; ms < 2; ++ms)
#pragma unroll
    for (int ns = 0; ns < 4; ++ns)
#pragma unroll
      for (int i = 0; i < 8; ++i) {
        const int row = Arow0 + wm * 32 + ms * 16 + i + ((lane >> 4) << 3);
        const int col = Ncol0 + wn * 64 + ns * 16 + (lane & 15);
        out[(size_t)row * HH + col] = tanh_fast(acc[ms * 4 + ns][i]);
      }
}

// ---------------------------------------------------------------------------
// Launch
//   ws layout: [0, 32*BT)              float score partial slices (4 MB)
//              [32*BT, 32*BT+256*2048) float [rx || hn] packed A   (2 MB)
// ---------------------------------------------------------------------------
extern "C" void kernel_launch(void* const* d_in, const int* in_sizes, int n_in,
                              void* d_out, int out_size, void* d_ws, size_t ws_size,
                              hipStream_t stream) {
  const float* h      = (const float*)d_in[0];
  const float* aspect = (const float*)d_in[1];
  const float* Wh     = (const float*)d_in[2];
  const float* Wv     = (const float*)d_in[3];
  const float* Wp     = (const float*)d_in[4];
  const float* Wx     = (const float*)d_in[5];
  const float* weight = (const float*)d_in[6];
  float* out = (float*)d_out;

  float* part = (float*)d_ws;                 // 32 * 32768 floats
  float* Araw = part + (size_t)32 * BT;       // 256 * 2048 floats

  k_scores<<<dim3(BT / 128, 16), 256, 0, stream>>>(h, aspect, Wh, Wv, weight, part);
  k_softmax_rx<<<BB, 256, 0, stream>>>(part, h, Araw);
  k_final<<<dim3(BB / 128, HH / 128), 256, 0, stream>>>(Araw, Wp, Wx, out);
}
